// BeatAlignmentModule_88948772700640
// MI455X (gfx1250) — matrix operand order repaired and checked by reading, not compiled, not run on hardware
//
#include <hip/hip_runtime.h>
#include <hip/hip_bf16.h>
#include <math.h>

typedef __bf16 bf16_t;
typedef __bf16 v16bf __attribute__((ext_vector_type(16)));
typedef float  v8f   __attribute__((ext_vector_type(8)));

#define TEMP1 4.0f
#define TEMP2 5.0f
#define TEMP3 10.0f
#define EPS_  1e-8f

constexpr int NB = 64;   // batch
constexpr int NL = 64;   // seq len
constexpr int ND = 512;  // dim
constexpr int SST = 65;  // padded f32 score stride (avoid bank conflicts on column pass)

// LDS: ecg_bf 64K + ecgT_bf 64K + sent_bf 64K + a2_bf 8K + Ssc 16.25K + 1K misc
constexpr size_t LDS_BYTES = (size_t)65536*3 + 8192 + 16640 + 1024;

// Load A fragment (16x32 bf16): lane<16 -> K {k0..k0+15},
// lane>=16 -> +16.  `arow` must already include the (hi?16:0) K-offset and the M row.
__device__ __forceinline__ v16bf load_afrag(const bf16_t* __restrict__ arow, int kb) {
    v16bf a;
    #pragma unroll
    for (int e = 0; e < 8; ++e) { a[e] = arow[kb + e]; a[e + 8] = arow[kb + 8 + e]; }
    return a;
}
// Load B fragment (32x16 bf16): lane n holds 16 contiguous K starting at k0 (+16 for hi half).
__device__ __forceinline__ v16bf load_bfrag(const bf16_t* __restrict__ brow, int kb) {
    v16bf b;
    #pragma unroll
    for (int e = 0; e < 16; ++e) { b[e] = brow[kb + e]; }
    return b;
}

__global__ void __launch_bounds__(256)
pair_kernel(const float* __restrict__ ecg, const float* __restrict__ sent,
            float* __restrict__ out, float* __restrict__ simbuf)
{
    const int j = blockIdx.x;   // ecg batch index
    const int i = blockIdx.y;   // sent batch index
    const int t = threadIdx.x;
    const int lane = t & 31;
    const int wave = t >> 5;

    extern __shared__ char smem_raw[];
    bf16_t* ecg_bf  = (bf16_t*)smem_raw;            // 64*512  (GEMM1 A, row-major s x d)
    bf16_t* ecgT_bf = ecg_bf  + NL * ND;            // 512*64  (GEMM2 B, d-major)
    bf16_t* sent_bf = ecgT_bf + ND * NL;            // 64*512  (GEMM1 B / GEMM2 dot)
    bf16_t* a2_bf   = sent_bf + NL * ND;            // 64*64   (GEMM2 A, row-major q x s)
    float*  Ssc     = (float*)(a2_bf + NL * NL);    // 64*65
    float*  sentN   = Ssc + NL * SST;               // 64
    float*  w12     = sentN + NL;                   // 64
    float*  w2sq    = w12 + NL;                     // 64
    float*  red     = w2sq + NL;                    // 64

    if (t < NL) { w12[t] = 0.f; w2sq[t] = 0.f; }

    // ---- load + f32->bf16 convert + transpose + sent row norms ----
    {
        const int row = t >> 2;            // 0..63
        const int c0  = (t & 3) * 128;     // quarter of D
        const float* eg = ecg + ((size_t)j * NL + row) * ND + c0;
        #pragma unroll 4
        for (int c = 0; c < 128; ++c) {
            float v = eg[c];
            bf16_t bv = (bf16_t)v;
            ecg_bf[row * ND + c0 + c]    = bv;
            ecgT_bf[(c0 + c) * NL + row] = bv;
        }
        const float* sg = sent + ((size_t)i * NL + row) * ND + c0;
        float ss = 0.f;
        #pragma unroll 4
        for (int c = 0; c < 128; ++c) {
            float v = sg[c];
            sent_bf[row * ND + c0 + c] = (bf16_t)v;
            ss += v * v;
        }
        ss += __shfl_xor(ss, 1);
        ss += __shfl_xor(ss, 2);
        if ((t & 3) == 0) sentN[row] = ss;
    }
    __syncthreads();

    // ---- GEMM1: S[s,q] = ecg[j,s,:] . sent[i,q,:]  (K=512) ----
    // 1x2 register blocking: each wave owns one (ts, q-pair) strip:
    //   ts = (wave>>1)*16, tq in { (wave&1)*32, (wave&1)*32+16 }.
    // One shared A fragment feeds two independent WMMA chains.
    {
        const int m   = lane & 15;
        const int hi  = lane >> 4;
        const int ts  = (wave >> 1) * 16;
        const int tq0 = (wave & 1) * 32;
        v8f acc0 = {}, acc1 = {};
        const bf16_t* arow  = ecg_bf  + (ts  + m) * ND + (hi ? 16 : 0);
        const bf16_t* brow0 = sent_bf + (tq0 + m) * ND + (hi ? 16 : 0);
        const bf16_t* brow1 = brow0 + 16 * ND;
        for (int kb = 0; kb < ND; kb += 32) {
            v16bf a  = load_afrag(arow, kb);
            v16bf b0 = load_bfrag(brow0, kb);
            v16bf b1 = load_bfrag(brow1, kb);
            acc0 = __builtin_amdgcn_wmma_f32_16x16x32_bf16(false, a, false, b0, (short)0, acc0, false, false);
            acc1 = __builtin_amdgcn_wmma_f32_16x16x32_bf16(false, a, false, b1, (short)0, acc1, false, false);
        }
        #pragma unroll
        for (int r = 0; r < 8; ++r) {
            Ssc[(ts + r + hi * 8) * SST + tq0 + m]      = acc0[r];
            Ssc[(ts + r + hi * 8) * SST + tq0 + 16 + m] = acc1[r];
        }
    }
    __syncthreads();

    // ---- softmax over q (rows of S); 4 threads per row ----
    {
        const int s  = t >> 2;
        float* rowp  = Ssc + s * SST + (t & 3) * 16;
        float mx = -INFINITY;
        #pragma unroll
        for (int c = 0; c < 16; ++c) mx = fmaxf(mx, rowp[c]);
        mx = fmaxf(mx, __shfl_xor(mx, 1));
        mx = fmaxf(mx, __shfl_xor(mx, 2));
        float sum = 0.f;
        #pragma unroll
        for (int c = 0; c < 16; ++c) { float e = __expf(rowp[c] - mx); rowp[c] = e; sum += e; }
        sum += __shfl_xor(sum, 1);
        sum += __shfl_xor(sum, 2);
        float inv = 1.f / sum;
        #pragma unroll
        for (int c = 0; c < 16; ++c) rowp[c] *= inv;
    }
    __syncthreads();

    // ---- softmax over s of TEMP1*attn1^T -> A2[q,s]; emit att_maps on diagonal ----
    {
        const int q  = t >> 2;
        const int s0 = (t & 3) * 16;
        float x[16];
        float mx = -INFINITY;
        #pragma unroll
        for (int c = 0; c < 16; ++c) { x[c] = TEMP1 * Ssc[(s0 + c) * SST + q]; mx = fmaxf(mx, x[c]); }
        mx = fmaxf(mx, __shfl_xor(mx, 1));
        mx = fmaxf(mx, __shfl_xor(mx, 2));
        float sum = 0.f;
        #pragma unroll
        for (int c = 0; c < 16; ++c) { x[c] = __expf(x[c] - mx); sum += x[c]; }
        sum += __shfl_xor(sum, 1);
        sum += __shfl_xor(sum, 2);
        float inv = 1.f / sum;
        float* amap = out + 1 + ((size_t)i * NL + q) * NL + s0;
        #pragma unroll
        for (int c = 0; c < 16; ++c) {
            float v = x[c] * inv;
            a2_bf[q * NL + s0 + c] = (bf16_t)v;
            if (i == j) amap[c] = v;
        }
    }
    __syncthreads();

    // ---- GEMM2: wctx[q,d] = sum_s A2[q,s]*ecg[j,s,d]; fold into w12 / |wctx|^2 ----
    // A2 fragments hoisted out of the d loop; d-tiles processed in independent pairs.
    {
        const int m  = lane & 15;
        const int hi = lane >> 4;
        for (int tq4 = 0; tq4 < 4; ++tq4) {
            const int q0 = tq4 * 16;
            const bf16_t* arow = a2_bf + (q0 + m) * NL + (hi ? 16 : 0);
            v16bf a0 = load_afrag(arow, 0);    // K = 0..31
            v16bf a1 = load_afrag(arow, 32);   // K = 32..63
            float acc12[8], acc22[8];
            #pragma unroll
            for (int r = 0; r < 8; ++r) { acc12[r] = 0.f; acc22[r] = 0.f; }
            #pragma unroll
            for (int p = 0; p < 2; ++p) {
                const int tdA = wave + p * 16;  // this wave's d-tiles: wave, wave+8, +16, +24
                const int tdB = tdA + 8;
                const bf16_t* browA = ecgT_bf + (tdA * 16 + m) * NL + (hi ? 16 : 0);
                const bf16_t* browB = ecgT_bf + (tdB * 16 + m) * NL + (hi ? 16 : 0);
                v16bf bA0 = load_bfrag(browA, 0);
                v16bf bA1 = load_bfrag(browA, 32);
                v16bf bB0 = load_bfrag(browB, 0);
                v16bf bB1 = load_bfrag(browB, 32);
                v8f accA = {}, accB = {};
                accA = __builtin_amdgcn_wmma_f32_16x16x32_bf16(false, a0, false, bA0, (short)0, accA, false, false);
                accB = __builtin_amdgcn_wmma_f32_16x16x32_bf16(false, a0, false, bB0, (short)0, accB, false, false);
                accA = __builtin_amdgcn_wmma_f32_16x16x32_bf16(false, a1, false, bA1, (short)0, accA, false, false);
                accB = __builtin_amdgcn_wmma_f32_16x16x32_bf16(false, a1, false, bB1, (short)0, accB, false, false);
                const int dA = tdA * 16 + m;
                const int dB = tdB * 16 + m;
                #pragma unroll
                for (int r = 0; r < 8; ++r) {
                    const int q = q0 + r + hi * 8;
                    float svA = (float)sent_bf[q * ND + dA];
                    float svB = (float)sent_bf[q * ND + dB];
                    acc12[r] += accA[r] * svA + accB[r] * svB;
                    acc22[r] += accA[r] * accA[r] + accB[r] * accB[r];
                }
            }
            #pragma unroll
            for (int r = 0; r < 8; ++r) {
                float a12 = acc12[r], a22 = acc22[r];
                a12 += __shfl_xor(a12, 1); a22 += __shfl_xor(a22, 1);
                a12 += __shfl_xor(a12, 2); a22 += __shfl_xor(a22, 2);
                a12 += __shfl_xor(a12, 4); a22 += __shfl_xor(a22, 4);
                a12 += __shfl_xor(a12, 8); a22 += __shfl_xor(a22, 8);
                if (m == 0) {
                    const int q = q0 + r + hi * 8;
                    atomicAdd(&w12[q],  a12);
                    atomicAdd(&w2sq[q], a22);
                }
            }
        }
    }
    __syncthreads();

    // ---- cosine, logsumexp over q -> sim[j,i] (reference transposes sim) ----
    if (t < NL) {
        float w1 = sqrtf(sentN[t]);
        float w2 = sqrtf(w2sq[t]);
        float cs = w12[t] / fmaxf(w1 * w2, EPS_);
        red[t] = __expf(TEMP2 * cs);
    }
    __syncthreads();
    if (t == 0) {
        float s = 0.f;
        #pragma unroll 8
        for (int q = 0; q < NL; ++q) s += red[q];
        simbuf[(size_t)j * NB + i] = TEMP3 * __logf(s);
    }
}

__global__ void __launch_bounds__(64)
loss_kernel(const float* __restrict__ sim, float* __restrict__ out)
{
    __shared__ float part[NB];
    const int r = threadIdx.x;
    float mx = -INFINITY;
    for (int c = 0; c < NB; ++c) mx = fmaxf(mx, sim[r * NB + c]);
    float s = 0.f;
    for (int c = 0; c < NB; ++c) s += __expf(sim[r * NB + c] - mx);
    float lse_row = mx + __logf(s);

    float mxc = -INFINITY;
    for (int c = 0; c < NB; ++c) mxc = fmaxf(mxc, sim[c * NB + r]);
    float sc = 0.f;
    for (int c = 0; c < NB; ++c) sc += __expf(sim[c * NB + r] - mxc);
    float lse_col = mxc + __logf(sc);

    float diag = sim[r * NB + r];
    part[r] = (lse_row - diag) + (lse_col - diag);
    __syncthreads();
    if (r == 0) {
        float tot = 0.f;
        for (int k = 0; k < NB; ++k) tot += part[k];
        out[0] = tot / (2.0f * (float)NB);
    }
}

extern "C" void kernel_launch(void* const* d_in, const int* in_sizes, int n_in,
                              void* d_out, int out_size, void* d_ws, size_t ws_size,
                              hipStream_t stream)
{
    const float* ecg  = (const float*)d_in[0];
    const float* sent = (const float*)d_in[1];
    float* out    = (float*)d_out;
    float* simbuf = (float*)d_ws;   // 64*64 f32 = 16 KB

    dim3 grid(NB, NB);              // block (j,i) -> one 64x64 pair
    pair_kernel<<<grid, dim3(256), LDS_BYTES, stream>>>(ecg, sent, out, simbuf);
    loss_kernel<<<dim3(1), dim3(64), 0, stream>>>(simbuf, out);
}
